// Transformer_81509889343612
// MI455X (gfx1250) — matrix operand order repaired and checked by reading, not compiled, not run on hardware
//
#include <hip/hip_runtime.h>
#include <cmath>

// ---------------------------------------------------------------------------
// CDNA5 (gfx1250) transformer forward.
// All GEMMs run through V_WMMA_F32_16X16X32_BF16 (bf16 in, fp32 accumulate).
// fp32 weights/activations are converted to bf16 on a register-staged,
// double-buffered LDS pipeline: issue global_load_b128 for tile kt+1,
// compute WMMAs for tile kt from LDS, then cvt+ds_store_b64 into the
// alternate buffer. One barrier per K-step. All trip counts constexpr.
// ---------------------------------------------------------------------------

typedef __bf16 bf16;
typedef __attribute__((ext_vector_type(4)))  __bf16 bf16x4;
typedef __attribute__((ext_vector_type(8)))  __bf16 bf16x8;
typedef __attribute__((ext_vector_type(16))) __bf16 v16bf;
typedef __attribute__((ext_vector_type(8)))  float  v8f;

#define LDT 40  // LDS row stride in bf16 elements (80B: 16B-aligned, padded)

// Generic batched GEMM:  C[z][m][n] = sum_k A[z][m][k] * B[z][n][k]  (+bias, gelu, +residual)
// Batch offsets: off(z) = (z / div) * s1 + (z % div) * s2   (covers head-sliced operands)
// BT = B operand stored transposed in memory: element (n,k) at offB + k*sBk + n  (sBn==1)
// KK = compile-time reduction length.
template<int WM, int WN, bool BT, int KK>
__global__ __launch_bounds__(WM*WN*32)
void gemm_wmma(const float* __restrict__ A, long sAm, int AbDiv, long sAb1, long sAb2,
               const float* __restrict__ Bp, long sBn, long sBk, int BbDiv, long sBb1, long sBb2,
               float* C, long sCm, int CbDiv, long sCb1, long sCb2,
               const float* __restrict__ bias, const float* resid, int gelu_flag)
{
    constexpr int BM = WM * 32, BN = WN * 32, BK = 32, NT = WM * WN * 32;
    constexpr int NKT = KK / BK;
    constexpr int IA  = (BM * (BK / 4)) / NT;              // A quads per thread
    constexpr int IB  = BT ? ((BN / 4) * BK) / NT          // B quads per thread
                           : (BN * (BK / 4)) / NT;
    constexpr int PR  = BN / 4;                            // quads per k-row (BT path)

    __shared__ __align__(16) bf16 lA[2][BM * LDT];
    __shared__ __align__(16) bf16 lB[2][BN * LDT];

    const int z = blockIdx.z;
    const long offA = (long)(z / AbDiv) * sAb1 + (long)(z % AbDiv) * sAb2;
    const long offB = (long)(z / BbDiv) * sBb1 + (long)(z % BbDiv) * sBb2;
    const long offC = (long)(z / CbDiv) * sCb1 + (long)(z % CbDiv) * sCb2;

    const int tileM = blockIdx.y * BM;
    const int tileN = blockIdx.x * BN;
    const int tid   = threadIdx.x;
    const int lane  = tid & 31;
    const int wave  = tid >> 5;
    const int wm = wave / WN, wn = wave % WN;

    float4 fa[IA];
    float4 fb[IB];

    // issue phase: global_load_b128 into registers (no LDS dependency -> loads pipeline)
    auto issue = [&](int k0) {
        #pragma unroll
        for (int it = 0; it < IA; ++it) {
            const int q = tid + it * NT;
            const int m = q >> 3, k4 = (q & 7) << 2;
            fa[it] = *(const float4*)(A + offA + (long)(tileM + m) * sAm + (k0 + k4));
        }
        if constexpr (!BT) {
            #pragma unroll
            for (int it = 0; it < IB; ++it) {
                const int q = tid + it * NT;
                const int n = q >> 3, k4 = (q & 7) << 2;
                fb[it] = *(const float4*)(Bp + offB + (long)(tileN + n) * sBn + (k0 + k4));
            }
        } else {
            #pragma unroll
            for (int it = 0; it < IB; ++it) {
                const int q = tid + it * NT;
                const int k = q / PR, n4 = (q % PR) << 2;
                fb[it] = *(const float4*)(Bp + offB + (long)(k0 + k) * sBk + (tileN + n4));
            }
        }
        if (k0 + BK < KK)   // gfx1250 global_prefetch_b8, two K-slabs ahead
            __builtin_prefetch(A + offA + (long)(tileM + (tid & (BM - 1))) * sAm + (k0 + BK), 0, 1);
    };
    // store phase: fp32 -> bf16 pack, ds_store_b64 (or b16 scatter for BT)
    auto stage = [&](int b) {
        #pragma unroll
        for (int it = 0; it < IA; ++it) {
            const int q = tid + it * NT;
            const int m = q >> 3, k4 = (q & 7) << 2;
            bf16x4 h;
            h[0] = (bf16)fa[it].x; h[1] = (bf16)fa[it].y;
            h[2] = (bf16)fa[it].z; h[3] = (bf16)fa[it].w;
            *(bf16x4*)(&lA[b][m * LDT + k4]) = h;
        }
        if constexpr (!BT) {
            #pragma unroll
            for (int it = 0; it < IB; ++it) {
                const int q = tid + it * NT;
                const int n = q >> 3, k4 = (q & 7) << 2;
                bf16x4 h;
                h[0] = (bf16)fb[it].x; h[1] = (bf16)fb[it].y;
                h[2] = (bf16)fb[it].z; h[3] = (bf16)fb[it].w;
                *(bf16x4*)(&lB[b][n * LDT + k4]) = h;
            }
        } else {
            #pragma unroll
            for (int it = 0; it < IB; ++it) {
                const int q = tid + it * NT;
                const int k = q / PR, n4 = (q % PR) << 2;
                lB[b][(n4 + 0) * LDT + k] = (bf16)fb[it].x;
                lB[b][(n4 + 1) * LDT + k] = (bf16)fb[it].y;
                lB[b][(n4 + 2) * LDT + k] = (bf16)fb[it].z;
                lB[b][(n4 + 3) * LDT + k] = (bf16)fb[it].w;
            }
        }
    };

    v8f acc[2][2] = {};

    issue(0);
    stage(0);
    for (int kt = 0; kt < NKT; ++kt) {
        __syncthreads();                       // buf[kt&1] stores visible; prior reads done
        if (kt + 1 < NKT) issue((kt + 1) * BK);

        // ---- fragments per ISA 7.12.2 VGPR layouts (wave32) ----
        const int cb = kt & 1;
        const int r  = lane & 15;
        const int hi = lane >> 4;
        v16bf af[2], bfr[2];
        #pragma unroll
        for (int i = 0; i < 2; ++i) {
            // A 16x32: lane r = row, elems 0..7 = K hb..hb+7, elems 8..15 = K hb+16..hb+23
            const bf16* p = &lA[cb][(wm * 32 + i * 16 + r) * LDT + hi * 8];
            bf16x8 c0 = *(const bf16x8*)(p);
            bf16x8 c1 = *(const bf16x8*)(p + 16);
            af[i] = __builtin_shufflevector(c0, c1, 0,1,2,3,4,5,6,7,8,9,10,11,12,13,14,15);
        }
        #pragma unroll
        for (int j = 0; j < 2; ++j) {
            // B 32x16: lane r = column, elems 0..7 = K hb..hb+7, elems 8..15 = K hb+16..hb+23
            const bf16* p = &lB[cb][(wn * 32 + j * 16 + r) * LDT + hi * 8];
            bf16x8 c0 = *(const bf16x8*)(p);
            bf16x8 c1 = *(const bf16x8*)(p + 16);
            bfr[j] = __builtin_shufflevector(c0, c1, 0,1,2,3,4,5,6,7,8,9,10,11,12,13,14,15);
        }
        #pragma unroll
        for (int i = 0; i < 2; ++i)
            #pragma unroll
            for (int j = 0; j < 2; ++j)
                acc[i][j] = __builtin_amdgcn_wmma_f32_16x16x32_bf16(
                    false, af[i], false, bfr[j], (short)0, acc[i][j], false, false);

        if (kt + 1 < NKT) stage((kt + 1) & 1);  // write other buffer; barrier next iter orders it
    }

    // ---- epilogue: C layout = 8 VGPRs, lanes 0-15 rows 0..7 / lanes 16-31 rows 8..15 ----
    const int r = lane & 15;
    const int hiOff = (lane >> 4) * 8;
    #pragma unroll
    for (int i = 0; i < 2; ++i) {
        const int rowBase = tileM + wm * 32 + i * 16 + hiOff;
        #pragma unroll
        for (int j = 0; j < 2; ++j) {
            const int gcol = tileN + wn * 32 + j * 16 + r;
            const float bb = bias ? bias[gcol] : 0.0f;
            #pragma unroll
            for (int e = 0; e < 8; ++e) {
                const int grow = rowBase + e;
                float v = acc[i][j][e] + bb;
                if (gelu_flag) v = 0.5f * v * (1.0f + erff(v * 0.70710678118654752f));
                const long ci = offC + (long)grow * sCm + gcol;
                if (resid) v += resid[ci];
                C[ci] = v;
            }
        }
    }
}

// LayerNorm over H=1024; optional fused residual (decoder post-norm: y = LN(x + res))
__global__ __launch_bounds__(256)
void ln_kernel(const float* __restrict__ x, const float* __restrict__ res,
               const float* __restrict__ g, const float* __restrict__ bta, float* __restrict__ y)
{
    const int row = blockIdx.x, tid = threadIdx.x;
    const long base = (long)row * 1024;
    float v[4];
    #pragma unroll
    for (int i = 0; i < 4; ++i) {
        const int c = tid + i * 256;
        float t = x[base + c];
        if (res) t += res[base + c];
        v[i] = t;
    }
    __shared__ float sd[256];
    sd[tid] = v[0] + v[1] + v[2] + v[3];
    __syncthreads();
    for (int st = 128; st > 0; st >>= 1) { if (tid < st) sd[tid] += sd[tid + st]; __syncthreads(); }
    const float mean = sd[0] * (1.0f / 1024.0f);
    __syncthreads();
    float q = 0.0f;
    #pragma unroll
    for (int i = 0; i < 4; ++i) { const float d = v[i] - mean; q += d * d; }
    sd[tid] = q;
    __syncthreads();
    for (int st = 128; st > 0; st >>= 1) { if (tid < st) sd[tid] += sd[tid + st]; __syncthreads(); }
    const float rstd = rsqrtf(sd[0] * (1.0f / 1024.0f) + 1e-5f);
    __syncthreads();
    #pragma unroll
    for (int i = 0; i < 4; ++i) {
        const int c = tid + i * 256;
        y[base + c] = (v[i] - mean) * rstd * g[c] + bta[c];
    }
}

// Softmax over rows of 96 (scaled by 1/sqrt(64), optional causal mask); 1 wave32 per row.
__global__ __launch_bounds__(256)
void softmax_kernel(float* __restrict__ sc, int causal)
{
    const int wave = threadIdx.x >> 5, lane = threadIdx.x & 31;
    const long row = (long)blockIdx.x * 8 + wave;      // 256 heads * 96 rows total
    const long z = row / 96;
    const int  mq = (int)(row % 96);
    float* p = sc + z * 9216 + (long)mq * 96;
    float v[3];
    #pragma unroll
    for (int t = 0; t < 3; ++t) {
        const int c = lane + t * 32;
        float val = p[c] * 0.125f;
        if (causal && c > mq) val = -3.0e38f;
        v[t] = val;
    }
    float mx = fmaxf(v[0], fmaxf(v[1], v[2]));
    for (int o = 16; o > 0; o >>= 1) mx = fmaxf(mx, __shfl_xor(mx, o, 32));
    float sum = 0.0f;
    #pragma unroll
    for (int t = 0; t < 3; ++t) { v[t] = __expf(v[t] - mx); sum += v[t]; }
    for (int o = 16; o > 0; o >>= 1) sum += __shfl_xor(sum, o, 32);
    const float inv = 1.0f / sum;
    #pragma unroll
    for (int t = 0; t < 3; ++t) p[lane + t * 32] = v[t] * inv;
}

// x (+ positional encoding) -> y, over (B, L=96, H=1024)
__global__ __launch_bounds__(256)
void addpos_kernel(const float* __restrict__ x, const float* __restrict__ pe,
                   float* __restrict__ y, long total)
{
    const long i = (long)blockIdx.x * 256 + threadIdx.x;
    if (i >= total) return;
    const int l = (int)((i >> 10) % 96);
    const int h = (int)(i & 1023);
    y[i] = x[i] + pe[l * 1024 + h];
}

extern "C" void kernel_launch(void* const* d_in, const int* in_sizes, int n_in,
                              void* d_out, int out_size, void* d_ws, size_t ws_size,
                              hipStream_t stream)
{
    (void)in_sizes; (void)n_in; (void)out_size; (void)ws_size;

    const float* src      = (const float*)d_in[0];
    const float* tgt      = (const float*)d_in[1];
    const float* pos      = (const float*)d_in[2];
    const float* enc_W    = (const float*)d_in[3];   // [6][4][1024][1024]
    const float* enc_W1   = (const float*)d_in[4];
    const float* enc_b1   = (const float*)d_in[5];
    const float* enc_W2   = (const float*)d_in[6];
    const float* enc_b2   = (const float*)d_in[7];
    const float* enc_ln_g = (const float*)d_in[8];   // [6][2][1024]
    const float* enc_ln_b = (const float*)d_in[9];
    const float* enc_fn_g = (const float*)d_in[10];
    const float* enc_fn_b = (const float*)d_in[11];
    const float* dec_Ws   = (const float*)d_in[12];  // [6][4][1024][1024]
    const float* dec_Wc   = (const float*)d_in[13];
    const float* dec_W1   = (const float*)d_in[14];
    const float* dec_b1   = (const float*)d_in[15];
    const float* dec_W2   = (const float*)d_in[16];
    const float* dec_b2   = (const float*)d_in[17];
    const float* dec_ln_g = (const float*)d_in[18];  // [6][3][1024]
    const float* dec_ln_b = (const float*)d_in[19];
    const float* dec_fn_g = (const float*)d_in[20];
    const float* dec_fn_b = (const float*)d_in[21];

    float* ws = (float*)d_ws;
    const size_t SZ = (size_t)1536 * 1024;          // one (B*L, H) activation buffer
    float* x_enc   = ws + 0 * SZ;
    float* x_dec   = ws + 1 * SZ;
    float* enc_out = ws + 2 * SZ;
    float* hbuf    = ws + 3 * SZ;
    float* qb      = ws + 4 * SZ;
    float* kb      = ws + 5 * SZ;
    float* vb      = ws + 6 * SZ;
    float* att     = ws + 7 * SZ;
    float* scores  = ws + 8 * SZ;                    // 256 * 96 * 96 floats

    // dense projection: C(1536x1024) = A(1536x1024) . W(1024x1024)^T   (K = 1024)
    auto proj = [&](const float* A, const float* W, float* Cc,
                    const float* bias, const float* res, int gelu) {
        dim3 g(16, 12, 1);                           // BN=64, BM=128
        gemm_wmma<4, 2, false, 1024><<<g, 256, 0, stream>>>(A, 1024, 1, 0, 0,
                                                            W, 1024, 1, 1, 0, 0,
                                                            Cc, 1024, 1, 0, 0,
                                                            bias, res, gelu);
    };
    // scores[z](96x96) = Qz(96x64) . Kz(96x64)^T,  z = b*16 + h  (K = 64)
    auto attn_scores = [&](const float* q, const float* k) {
        dim3 g(3, 3, 256);
        gemm_wmma<1, 1, false, 64><<<g, 32, 0, stream>>>(q, 1024, 16, 98304, 64,
                                                         k, 1024, 1, 16, 98304, 64,
                                                         scores, 96, 1, 9216, 0,
                                                         nullptr, nullptr, 0);
    };
    // ctx[z](96x64) = P[z](96x96) . V[z](96x64)  (B transposed in memory, K = 96)
    auto attn_ctx = [&](const float* v, float* o) {
        dim3 g(2, 3, 256);
        gemm_wmma<1, 1, true, 96><<<g, 32, 0, stream>>>(scores, 96, 1, 9216, 0,
                                                        v, 1, 1024, 16, 98304, 64,
                                                        o, 1024, 16, 98304, 64,
                                                        nullptr, nullptr, 0);
    };
    auto ln = [&](const float* x, const float* res, const float* g_, const float* b_, float* y) {
        ln_kernel<<<1536, 256, 0, stream>>>(x, res, g_, b_, y);
    };
    auto softmax = [&](int causal) {
        softmax_kernel<<<3072, 256, 0, stream>>>(scores, causal);
    };

    const long total = (long)1536 * 1024;
    const int pb = (int)((total + 255) / 256);
    addpos_kernel<<<pb, 256, 0, stream>>>(src, pos, x_enc, total);
    addpos_kernel<<<pb, 256, 0, stream>>>(tgt, pos, x_dec, total);

    const size_t WMAT = (size_t)1024 * 1024;

    // ---- encoder (pre-norm) ----
    for (int i = 0; i < 6; ++i) {
        const float* W = enc_W + (size_t)i * 4 * WMAT;
        ln(x_enc, nullptr, enc_ln_g + (i * 2 + 0) * 1024, enc_ln_b + (i * 2 + 0) * 1024, hbuf);
        proj(hbuf, W + 0 * WMAT, qb, nullptr, nullptr, 0);
        proj(hbuf, W + 1 * WMAT, kb, nullptr, nullptr, 0);
        proj(hbuf, W + 2 * WMAT, vb, nullptr, nullptr, 0);
        attn_scores(qb, kb);
        softmax(0);
        attn_ctx(vb, att);
        proj(att, W + 3 * WMAT, x_enc, nullptr, x_enc, 0);          // residual add
        ln(x_enc, nullptr, enc_ln_g + (i * 2 + 1) * 1024, enc_ln_b + (i * 2 + 1) * 1024, hbuf);
        proj(hbuf, enc_W1 + i * WMAT, qb, enc_b1 + i * 1024, nullptr, 1);   // GELU
        proj(qb, enc_W2 + i * WMAT, x_enc, enc_b2 + i * 1024, x_enc, 0);    // bias + residual
    }
    ln(x_enc, nullptr, enc_fn_g, enc_fn_b, enc_out);

    // ---- decoder (post-norm) ----
    for (int i = 0; i < 6; ++i) {
        const float* Wsl = dec_Ws + (size_t)i * 4 * WMAT;
        const float* Wcl = dec_Wc + (size_t)i * 4 * WMAT;
        // causal self-attention
        proj(x_dec, Wsl + 0 * WMAT, qb, nullptr, nullptr, 0);
        proj(x_dec, Wsl + 1 * WMAT, kb, nullptr, nullptr, 0);
        proj(x_dec, Wsl + 2 * WMAT, vb, nullptr, nullptr, 0);
        attn_scores(qb, kb);
        softmax(1);
        attn_ctx(vb, att);
        proj(att, Wsl + 3 * WMAT, hbuf, nullptr, nullptr, 0);
        ln(x_dec, hbuf, dec_ln_g + (i * 3 + 0) * 1024, dec_ln_b + (i * 3 + 0) * 1024, x_dec);
        // cross-attention (KV from encoder output)
        proj(x_dec,   Wcl + 0 * WMAT, qb, nullptr, nullptr, 0);
        proj(enc_out, Wcl + 1 * WMAT, kb, nullptr, nullptr, 0);
        proj(enc_out, Wcl + 2 * WMAT, vb, nullptr, nullptr, 0);
        attn_scores(qb, kb);
        softmax(0);
        attn_ctx(vb, att);
        proj(att, Wcl + 3 * WMAT, hbuf, nullptr, nullptr, 0);
        ln(x_dec, hbuf, dec_ln_g + (i * 3 + 1) * 1024, dec_ln_b + (i * 3 + 1) * 1024, x_dec);
        // MLP
        proj(x_dec, dec_W1 + i * WMAT, qb, dec_b1 + i * 1024, nullptr, 1);
        proj(qb,    dec_W2 + i * WMAT, hbuf, dec_b2 + i * 1024, nullptr, 0);
        ln(x_dec, hbuf, dec_ln_g + (i * 3 + 2) * 1024, dec_ln_b + (i * 3 + 2) * 1024, x_dec);
    }
    ln(x_dec, nullptr, dec_fn_g, dec_fn_b, (float*)d_out);
}
